// BnbQloraMLP_712964571613
// MI455X (gfx1250) — compile-verified, matrix operand order checked by reading, not run on hardware
//
#include <hip/hip_runtime.h>
#include <hip/hip_bf16.h>

// ---------------------------------------------------------------------------
// Types for CDNA5 WMMA (wave32, 16x16x32 bf16 -> f32)
// ---------------------------------------------------------------------------
typedef __attribute__((ext_vector_type(16))) __bf16 v16bf;
typedef __attribute__((ext_vector_type(8)))  __bf16 v8bf;
typedef __attribute__((ext_vector_type(8)))  float  v8f;

// NF4 codebook (bitsandbytes)
__device__ __constant__ float NF4_CODE_D[16] = {
    -1.0f, -0.6961928009986877f, -0.5250730514526367f, -0.39491748809814453f,
    -0.28444138169288635f, -0.18477343022823334f, -0.09105003625154495f, 0.0f,
    0.07958029955625534f, 0.16093020141124725f, 0.24611230194568634f,
    0.33791524171829224f, 0.44070982933044434f, 0.5626170039176941f,
    0.7229568362236023f, 1.0f};

__device__ inline unsigned f32_bf16_bits(float f) {
  unsigned u = __float_as_uint(f);
  return (u + 0x7FFFu + ((u >> 16) & 1u)) >> 16;  // round-to-nearest-even
}
__device__ inline __bf16 f32_bf16(float f) {
  unsigned short s = (unsigned short)f32_bf16_bits(f);
  return __builtin_bit_cast(__bf16, s);
}

// Low 32 bits of a generic pointer to LDS == wave-relative LDS byte offset
// (addrspacecast AS3->generic places the LDS offset in the low dword).
__device__ inline unsigned lds_addr32(const void* p) {
  return (unsigned)(unsigned long long)(uintptr_t)p;
}

// 4 x b128 async copies global -> LDS (GVS mode: SGPR base + per-lane byte
// offset; INST_OFFSET is added to BOTH the global and the LDS address, so one
// (lds,goff) pair covers a 64-byte chunk). Tracked by ASYNCcnt.
__device__ inline void async_cp64B(unsigned lds, unsigned goff,
                                   const void* base) {
  asm volatile(
      "global_load_async_to_lds_b128 %0, %1, %2 offset:0\n"
      "global_load_async_to_lds_b128 %0, %1, %2 offset:16\n"
      "global_load_async_to_lds_b128 %0, %1, %2 offset:32\n"
      "global_load_async_to_lds_b128 %0, %1, %2 offset:48\n"
      :
      : "v"(lds), "v"(goff), "s"(base)
      : "memory");
}
__device__ inline void wait_async0() {
  asm volatile("s_wait_asynccnt 0x0" ::: "memory");
}

// ---------------------------------------------------------------------------
// NF4 quant->dequant: one wave32 per 64-element block (2 elems/lane).
// ---------------------------------------------------------------------------
__global__ __launch_bounds__(256)
void nf4_dequant_bf16(const float* __restrict__ w, __bf16* __restrict__ o,
                      int nblocks) {
  int blk  = blockIdx.x * (blockDim.x >> 5) + (threadIdx.x >> 5);
  if (blk >= nblocks) return;
  int lane = threadIdx.x & 31;
  size_t base = (size_t)blk * 64;

  float v0 = w[base + lane];
  float v1 = w[base + 32 + lane];
  float am = fmaxf(fabsf(v0), fabsf(v1));
#pragma unroll
  for (int m = 16; m >= 1; m >>= 1) am = fmaxf(am, __shfl_xor(am, m, 32));
  float inv = (am == 0.f) ? 1.f : (1.f / am);

  auto qd = [&](float v) -> float {
    float nrm = v * inv;
    int idx = 0;
#pragma unroll
    for (int i = 0; i < 15; ++i)
      idx += (0.5f * (NF4_CODE_D[i] + NF4_CODE_D[i + 1]) < nrm) ? 1 : 0;
    return NF4_CODE_D[idx] * am;
  };
  o[base + lane]      = f32_bf16(qd(v0));
  o[base + 32 + lane] = f32_bf16(qd(v1));
}

// ---------------------------------------------------------------------------
// x: fp32 -> bf16, 4 elements per thread
// ---------------------------------------------------------------------------
__global__ __launch_bounds__(256)
void cvt_f32_bf16_x4(const float4* __restrict__ in,
                     unsigned long long* __restrict__ o, int n4) {
  int i = blockIdx.x * 256 + threadIdx.x;
  if (i >= n4) return;
  float4 v = in[i];
  unsigned long long p =
      (unsigned long long)f32_bf16_bits(v.x) |
      ((unsigned long long)f32_bf16_bits(v.y) << 16) |
      ((unsigned long long)f32_bf16_bits(v.z) << 32) |
      ((unsigned long long)f32_bf16_bits(v.w) << 48);
  o[i] = p;
}

// ---------------------------------------------------------------------------
// 16-bit WMMA fragment per ISA 16x32 layout:
//   lanes 0-15 : VGPR0-3 = K 0..7 , VGPR4-7 = K 16..23   (ch = 0)
//   lanes16-31 : VGPR0-3 = K 8..15, VGPR4-7 = K 24..31   (ch = 8)
// ---------------------------------------------------------------------------
__device__ inline v16bf ld_frag(const __bf16* row, int ch) {
  v8bf lo = *(const v8bf*)(row + ch);
  v8bf hi = *(const v8bf*)(row + 16 + ch);
  v16bf r;
#pragma unroll
  for (int i = 0; i < 8; ++i) { r[i] = lo[i]; r[i + 8] = hi[i]; }
  return r;
}

// ---------------------------------------------------------------------------
// Async double-buffered LDS-tiled WMMA GEMM:  out[M,N] = A[M,K] * W[N,K]^T
// 256 threads / 8 waves; tile 128(M) x 64(N); K-stage 64 (2 wmma K-steps).
// Wave (wy 0..3, wx 0..1) owns 32x32 = 2x2 wmma tiles.
// Pipeline (unrolled x2 so buffer index is compile-time):
//   wait_async -> barrier -> issue next stage -> compute current.
// Requires an even number of 64-wide K stages (holds: 4096/64, 11008/64).
// FUSED: gate+up against B1/B2, epilogue silu(g)*u -> bf16.
// ---------------------------------------------------------------------------
template <bool FUSED>
__global__ __launch_bounds__(256)
void wmma_gemm(const __bf16* __restrict__ A, const __bf16* __restrict__ B1,
               const __bf16* __restrict__ B2, __bf16* __restrict__ outH,
               float* __restrict__ outF, int M, int N, int K) {
  __shared__ __align__(16) __bf16 sA[2][128 * 64];
  __shared__ __align__(16) __bf16 sB1[2][64 * 64];
  __shared__ __align__(16) __bf16 sB2[FUSED ? 2 : 1][FUSED ? 64 * 64 : 16];

  const int tid  = threadIdx.x;
  const int lane = tid & 31;
  const int wave = tid >> 5;
  const int wy   = wave >> 1;  // 0..3 -> M slice
  const int wx   = wave & 1;   // 0..1 -> N slice
  const int tileM = blockIdx.y * 128;
  const int tileN = blockIdx.x * 64;

  // async copy assignments: 64 bytes (32 halfs) per thread per tile
  const int ar = tid >> 1;           // A row 0..127
  const int ac = (tid & 1) * 32;     // half offset in 64-half row
  const int br = (tid & 127) >> 1;   // B row 0..63
  const int bc = (tid & 1) * 32;

  unsigned aLds[2], b1Lds[2], b2Lds[2];
#pragma unroll
  for (int b = 0; b < 2; ++b) {
    aLds[b]  = lds_addr32(&sA[b][ar * 64 + ac]);
    b1Lds[b] = lds_addr32(&sB1[b][br * 64 + bc]);
    if constexpr (FUSED) b2Lds[b] = lds_addr32(&sB2[b][br * 64 + bc]);
  }
  const unsigned aOff  = (unsigned)(((size_t)(tileM + ar) * K + ac) * 2);
  const unsigned b1Off = (unsigned)(((size_t)(tileN + br) * K + bc) * 2);
  unsigned b2Off = 0;
  if constexpr (FUSED) b2Off = (unsigned)(((size_t)(tileN + br) * K + bc) * 2);

  auto issue = [&](int k0, int buf) {
    unsigned koff = (unsigned)k0 * 2u;  // bytes
    async_cp64B(aLds[buf], aOff + koff, A);
    if (tid < 128) {
      async_cp64B(b1Lds[buf], b1Off + koff, B1);
    } else {
      if constexpr (FUSED) async_cp64B(b2Lds[buf], b2Off + koff, B2);
    }
  };

  const int ch = (lane >> 4) << 3;  // 0 or 8
  const int fr = lane & 15;

  v8f accG[2][2] = {};
  v8f accU[2][2] = {};

  auto stage_compute = [&](const __bf16* bA, const __bf16* bB1,
                           const __bf16* bB2) {
#pragma unroll
    for (int kk = 0; kk < 64; kk += 32) {
      v16bf a0 = ld_frag(bA + (wy * 32 + fr) * 64 + kk, ch);
      v16bf a1 = ld_frag(bA + (wy * 32 + 16 + fr) * 64 + kk, ch);
      v16bf p0 = ld_frag(bB1 + (wx * 32 + fr) * 64 + kk, ch);
      v16bf p1 = ld_frag(bB1 + (wx * 32 + 16 + fr) * 64 + kk, ch);

      accG[0][0] = __builtin_amdgcn_wmma_f32_16x16x32_bf16(
          false, a0, false, p0, (short)0, accG[0][0], false, false);
      accG[0][1] = __builtin_amdgcn_wmma_f32_16x16x32_bf16(
          false, a0, false, p1, (short)0, accG[0][1], false, false);
      accG[1][0] = __builtin_amdgcn_wmma_f32_16x16x32_bf16(
          false, a1, false, p0, (short)0, accG[1][0], false, false);
      accG[1][1] = __builtin_amdgcn_wmma_f32_16x16x32_bf16(
          false, a1, false, p1, (short)0, accG[1][1], false, false);

      if constexpr (FUSED) {
        v16bf q0 = ld_frag(bB2 + (wx * 32 + fr) * 64 + kk, ch);
        v16bf q1 = ld_frag(bB2 + (wx * 32 + 16 + fr) * 64 + kk, ch);
        accU[0][0] = __builtin_amdgcn_wmma_f32_16x16x32_bf16(
            false, a0, false, q0, (short)0, accU[0][0], false, false);
        accU[0][1] = __builtin_amdgcn_wmma_f32_16x16x32_bf16(
            false, a0, false, q1, (short)0, accU[0][1], false, false);
        accU[1][0] = __builtin_amdgcn_wmma_f32_16x16x32_bf16(
            false, a1, false, q0, (short)0, accU[1][0], false, false);
        accU[1][1] = __builtin_amdgcn_wmma_f32_16x16x32_bf16(
            false, a1, false, q1, (short)0, accU[1][1], false, false);
      }
    }
  };

  const int nst = K >> 6;  // 64-wide K stages; even for all our shapes
  issue(0, 0);
  for (int s = 0; s < nst; s += 2) {
    // ---- stage s from buffer 0 ----
    wait_async0();     // my async writes into buf0 landed in LDS
    __syncthreads();   // everyone's landed; buf1 fully consumed
    if (s + 1 < nst) issue((s + 1) << 6, 1);
    stage_compute(sA[0], sB1[0], sB2[0]);
    // ---- stage s+1 from buffer 1 ----
    wait_async0();
    __syncthreads();
    if (s + 2 < nst) issue((s + 2) << 6, 0);
    stage_compute(sA[1], sB1[1], sB2[FUSED ? 1 : 0]);
  }

  // Epilogue. D layout: lanes 0-15 hold N=fr rows M=r; lanes 16-31 rows M=r+8.
#pragma unroll
  for (int mi = 0; mi < 2; ++mi) {
#pragma unroll
    for (int ni = 0; ni < 2; ++ni) {
      int col = tileN + wx * 32 + ni * 16 + fr;
      int rb  = tileM + wy * 32 + mi * 16 + ((lane >> 4) << 3);
#pragma unroll
      for (int r = 0; r < 8; ++r) {
        if constexpr (FUSED) {
          float g = accG[mi][ni][r];
          float u = accU[mi][ni][r];
          float sl = g / (1.f + __expf(-g));  // silu
          outH[(size_t)(rb + r) * N + col] = f32_bf16(sl * u);
        } else {
          outF[(size_t)(rb + r) * N + col] = accG[mi][ni][r];
        }
      }
    }
  }
}

// ---------------------------------------------------------------------------
// Launcher
// ---------------------------------------------------------------------------
extern "C" void kernel_launch(void* const* d_in, const int* in_sizes, int n_in,
                              void* d_out, int out_size, void* d_ws,
                              size_t ws_size, hipStream_t stream) {
  const float* x  = (const float*)d_in[0];
  const float* w1 = (const float*)d_in[1];
  const float* w2 = (const float*)d_in[2];
  const float* w3 = (const float*)d_in[3];
  float* out = (float*)d_out;

  const int D = 4096, H = 11008;
  const int M = in_sizes[0] / D;  // 2*2048 = 4096 tokens

  char* ws = (char*)d_ws;
  size_t off = 0;
  auto carve = [&](size_t bytes) -> char* {
    char* p = ws + off;
    off += (bytes + 255) & ~(size_t)255;
    return p;
  };
  __bf16* xb   = (__bf16*)carve((size_t)M * D * 2);
  __bf16* dq1  = (__bf16*)carve((size_t)H * D * 2);
  __bf16* dq2  = (__bf16*)carve((size_t)H * D * 2);
  __bf16* dq3  = (__bf16*)carve((size_t)D * H * 2);
  __bf16* hbuf = (__bf16*)carve((size_t)M * H * 2);

  // 1) NF4 quant->dequant all three weights to bf16 (8 blocks-of-64 per WG)
  int nbw = (H * D) / 64;
  dim3 dq_grid((nbw + 7) / 8);
  nf4_dequant_bf16<<<dq_grid, 256, 0, stream>>>(w1, dq1, nbw);
  nf4_dequant_bf16<<<dq_grid, 256, 0, stream>>>(w2, dq2, nbw);
  nf4_dequant_bf16<<<dq_grid, 256, 0, stream>>>(w3, dq3, nbw);

  // 2) x -> bf16
  int n4 = (M * D) / 4;
  cvt_f32_bf16_x4<<<(n4 + 255) / 256, 256, 0, stream>>>(
      (const float4*)x, (unsigned long long*)xb, n4);

  // 3) fused gate/up GEMM + SwiGLU: h = silu(x@w1^T) * (x@w2^T)   [M,H] bf16
  wmma_gemm<true><<<dim3(H / 64, M / 128), 256, 0, stream>>>(
      xb, dq1, dq2, hbuf, nullptr, M, H, D);

  // 4) out = h @ w3^T   [M,D] fp32
  wmma_gemm<false><<<dim3(D / 64, M / 128), 256, 0, stream>>>(
      hbuf, dq3, nullptr, nullptr, out, M, D, H);
}
